// SphericalVAE_84567906058761
// MI455X (gfx1250) — matrix operand order repaired and checked by reading, not compiled, not run on hardware
//
#include <hip/hip_runtime.h>
#include <hip/hip_bf16.h>
#include <stdint.h>

typedef __attribute__((ext_vector_type(16))) _Float16     v16h;
typedef __attribute__((ext_vector_type(8)))  _Float16     v8h;
typedef __attribute__((ext_vector_type(8)))  float        v8f;
typedef __attribute__((ext_vector_type(4)))  unsigned int u32x4;
typedef __attribute__((ext_vector_type(8)))  int          i32x8;
typedef __attribute__((ext_vector_type(4)))  int          i32x4;

#define BN_EPS 1e-5f
#define KMAX   1152   // largest Cin*9 (convT1: 128*9)

#ifndef __has_builtin
#define __has_builtin(x) 0
#endif
#if __has_builtin(__builtin_amdgcn_tensor_load_to_lds)
#define HAVE_TDM 1
#else
#define HAVE_TDM 0
#endif

__device__ __forceinline__ float lrelu(float x)    { return x >= 0.f ? x : 0.2f * x; }
__device__ __forceinline__ float sigmoidf(float x) { return 1.f / (1.f + expf(-x)); }

__device__ __forceinline__ void tdm_load(u32x4 g0, i32x8 g1, i32x4 g2, i32x4 g3) {
#if HAVE_TDM
#if __clang_major__ >= 23
  i32x8 g4 = {0, 0, 0, 0, 0, 0, 0, 0};
  __builtin_amdgcn_tensor_load_to_lds(g0, g1, g2, g3, g4, 0);
#else
  __builtin_amdgcn_tensor_load_to_lds(g0, g1, g2, g3, 0);
#endif
#endif
}

__device__ __forceinline__ void wait_tensor0() {
#if __has_builtin(__builtin_amdgcn_s_wait_tensorcnt)
  __builtin_amdgcn_s_wait_tensorcnt(0);
#else
  asm volatile("s_wait_tensorcnt 0x0" ::: "memory");
#endif
}

// ---------------------------------------------------------------------------
// f32 -> f16 cast
// ---------------------------------------------------------------------------
__global__ void cast_h_k(const float* __restrict__ in, _Float16* __restrict__ out, int n) {
  int i = blockIdx.x * blockDim.x + threadIdx.x;
  if (i < n) out[i] = (_Float16)in[i];
}

// ---------------------------------------------------------------------------
// Implicit-GEMM conv / convT (k=3, s=2, p=1 [+outpad 1 for convT]).
//   GEMM: M = Cout, N = B*Hout*Wout, K = Cin*9 (zero-padded to 32).
//   Block = 256 thr (8 waves); tile 16(M) x 256(N); each wave: two 16x16
//   sub-tiles -> 2 v_wmma per K-step reusing one A fragment.
//   A tile staged once per block: TDM tensor_load_to_lds (async, TENSORcnt)
//   or VALU fallback (conv1: K=27 breaks TDM 4B tile_dim0 rule).
//   B tile [N=256][K=32] (f16, +40-half row pitch): spatial decode hoisted
//   (nn == tid), register-gathered, 4x ds_store_b128 per thread; fragment =
//   two aligned 16B LDS vector loads (K-contiguous halves).
//   All dims power-of-two: osh = log2(Wout)=log2(Hout), ish = log2(Win).
// ---------------------------------------------------------------------------
template <int TRANSPOSED, int ACT_SIGMOID, int USE_TDM>
__global__ void __launch_bounds__(256)
conv_wmma_k(const _Float16* __restrict__ in, const _Float16* __restrict__ w,
            const float* __restrict__ bias,
            const float* __restrict__ bg, const float* __restrict__ bb,
            const float* __restrict__ bm, const float* __restrict__ bv,
            _Float16* __restrict__ out_h, float* __restrict__ out_f,
            int Cin, int Cout, int osh, int ish) {
  const int KTOT = Cin * 9;
  const int KPAD = (KTOT + 31) & ~31;
  const int Wout = 1 << osh, Hout = Wout;
  const int Win  = 1 << ish, Hin  = Win;

  __shared__ _Float16 Ash[16 * KMAX];      // [m][k], row stride = KTOT
  __shared__ _Float16 Bst[256][40];        // [n][k], 80B pitch (16B-aligned)

  const int tid  = threadIdx.x;
  const int lane = tid & 31;
  const int wv   = tid >> 5;
  const int ml   = lane & 15;
  const int hi   = lane >> 4;
  const int m0   = blockIdx.y * 16;
  const int n0   = blockIdx.x * 256;

  // ---- per-thread spatial decode (hoisted: nn == tid) ---------------------
  const int n  = n0 + tid;
  const int ox = n & (Wout - 1);
  const int oy = (n >> osh) & (Hout - 1);
  const int b  = n >> (2 * osh);
  const int oy2 = TRANSPOSED ? (oy + 1) : (oy * 2 - 1);  // iy = oy2 - ky (T) / + ky
  const int ox2 = TRANSPOSED ? (ox + 1) : (ox * 2 - 1);

  // ---- stage A (whole 16 x KTOT weight tile) ------------------------------
  if (USE_TDM && HAVE_TDM) {
    if (tid < 32) {                        // one TDM issue per block (wave 0)
      const _Float16* wt;
      uint32_t td0, td1, td2, tl0, tl1, tl2;
      uint64_t s0, s1;
      if (!TRANSPOSED) {                   // w: (Cout, Cin*9) row-major
        wt  = w + (size_t)m0 * KTOT;
        td0 = (uint32_t)KTOT; td1 = (uint32_t)(Cout - m0); td2 = 0;
        tl0 = (uint32_t)KTOT; tl1 = 16;    tl2 = 0;
        s0  = (uint64_t)KTOT; s1 = 0;
      } else {                             // w: (Cin, Cout, 9); A[m][ci*9+r]
        wt  = w + (size_t)m0 * 9;
        td0 = 9; td1 = (uint32_t)Cin; td2 = (uint32_t)(Cout - m0);
        tl0 = 9; tl1 = (uint32_t)Cin; tl2 = 16;
        s0  = (uint64_t)Cout * 9;     s1  = 9;
      }
      uint64_t ga  = (uint64_t)(uintptr_t)wt;
      uint32_t lds = (uint32_t)(uintptr_t)&Ash[0];
      u32x4 g0; i32x8 g1; i32x4 g2, g3;
      g0[0] = 1u;                                   // count=1, user mode
      g0[1] = lds;
      g0[2] = (uint32_t)ga;
      g0[3] = (uint32_t)((ga >> 32) & 0x1ffffffu) | (2u << 30); // type=2
      g1[0] = (int)(1u << 16);                      // data_size=1 (2 bytes)
      g1[1] = (int)((td0 & 0xffffu) << 16);
      g1[2] = (int)((td0 >> 16) | ((td1 & 0xffffu) << 16));
      g1[3] = (int)((td1 >> 16) | (tl0 << 16));
      g1[4] = (int)(tl1 | (tl2 << 16));
      g1[5] = (int)(uint32_t)(s0 & 0xffffffffu);
      g1[6] = (int)((uint32_t)(s0 >> 32) | ((uint32_t)(s1 & 0xffffu) << 16));
      g1[7] = (int)(uint32_t)(s1 >> 16);
      g2[0] = (int)td2; g2[1] = 0; g2[2] = 0; g2[3] = 0;
      g3[0] = 0; g3[1] = 0; g3[2] = 0; g3[3] = 0;
      tdm_load(g0, g1, g2, g3);
    }
  } else {
    #pragma unroll 1
    for (int m = 0; m < 16; ++m) {
      for (int k = tid; k < KTOT; k += 256) {
        _Float16 v = (_Float16)0.f;
        if (m0 + m < Cout) {
          if (!TRANSPOSED) {
            v = w[(size_t)(m0 + m) * KTOT + k];
          } else {
            int ci = k / 9, r = k - ci * 9;
            v = w[(size_t)(ci * Cout + m0 + m) * 9 + r];
          }
        }
        Ash[m * KTOT + k] = v;
      }
    }
  }

  v8f acc0 = {}, acc1 = {};

  for (int ks = 0; ks < KPAD; ks += 32) {
    // --- gather this thread's B row (32 halves) into registers ------------
    v8h chunk[4];
    #pragma unroll
    for (int t = 0; t < 32; ++t) {
      int k = ks + t;
      _Float16 v = (_Float16)0.f;
      if (k < KTOT) {
        int ci = k / 9; int r = k - ci * 9; int ky = r / 3, kx = r - ky * 3;
        if (!TRANSPOSED) {
          int iy = oy2 + ky, ix = ox2 + kx;
          if (iy >= 0 && iy < Hin && ix >= 0 && ix < Win)
            v = in[((b * Cin + ci) << (2 * ish)) + (iy << ish) + ix];
        } else {
          int ty = oy2 - ky, tx = ox2 - kx;         // inverted stride-2 scatter
          if (ty >= 0 && tx >= 0 && !(ty & 1) && !(tx & 1)) {
            int iy = ty >> 1, ix = tx >> 1;
            if (iy < Hin && ix < Win)
              v = in[((b * Cin + ci) << (2 * ish)) + (iy << ish) + ix];
          }
        }
      }
      chunk[t >> 3][t & 7] = v;
    }
    if (ks == 0 && USE_TDM && HAVE_TDM) {
      if (tid < 32) wait_tensor0();        // A tile landed (overlapped w/ gather)
    }
    // previous iteration's fragment reads completed at trailing barrier
    v8h* brow = (v8h*)&Bst[tid][0];
    brow[0] = chunk[0]; brow[1] = chunk[1]; brow[2] = chunk[2]; brow[3] = chunk[3];
    __syncthreads();

    // --- fragments; A reused across two WMMAs -----------------------------
    v16h af;
    #pragma unroll
    for (int v = 0; v < 8; ++v) {
      int ka = hi * 16 + 2 * v;
      af[2 * v]     = Ash[ml * KTOT + ks + ka];     // B side zero for k>=KTOT
      af[2 * v + 1] = Ash[ml * KTOT + ks + ka + 1];
    }
    const _Float16* br0 = &Bst[wv * 32 + ml][hi * 16];
    const _Float16* br1 = &Bst[wv * 32 + 16 + ml][hi * 16];
    v8h l0 = *(const v8h*)br0, h0 = *(const v8h*)(br0 + 8);
    v8h l1 = *(const v8h*)br1, h1 = *(const v8h*)(br1 + 8);
    v16h bf0 = __builtin_shufflevector(l0, h0, 0,1,2,3,4,5,6,7,8,9,10,11,12,13,14,15);
    v16h bf1 = __builtin_shufflevector(l1, h1, 0,1,2,3,4,5,6,7,8,9,10,11,12,13,14,15);

    acc0 = __builtin_amdgcn_wmma_f32_16x16x32_f16(false, af, false, bf0,
                                                  (short)0, acc0, false, false);
    acc1 = __builtin_amdgcn_wmma_f32_16x16x32_f16(false, af, false, bf1,
                                                  (short)0, acc1, false, false);
    __syncthreads();
  }

  // --- epilogue: bias + BN + lrelu (f16) or bias + sigmoid (f32) -----------
  #pragma unroll
  for (int sub = 0; sub < 2; ++sub) {
    const v8f& acc = sub ? acc1 : acc0;
    const int ne = n0 + wv * 32 + sub * 16 + ml;
    int oxe = ne & (Wout - 1);
    int oye = (ne >> osh) & (Hout - 1);
    int be  = ne >> (2 * osh);
    #pragma unroll
    for (int r = 0; r < 8; ++r) {
      int m = m0 + r + hi * 8;             // C/D layout: VGPR r -> M=r (+8 hi half)
      if (m < Cout) {
        float v = acc[r] + bias[m];
        int oidx = ((be * Cout + m) << (2 * osh)) + (oye << osh) + oxe;
        if (!ACT_SIGMOID) {
          float sc = bg[m] * rsqrtf(bv[m] + BN_EPS);
          v = lrelu((v - bm[m]) * sc + bb[m]);
          out_h[oidx] = (_Float16)v;
        } else {
          out_f[oidx] = sigmoidf(v);
        }
      }
    }
  }
}

// ---------------------------------------------------------------------------
// FC GEMM: one wave per 16x16 D tile. D[M,N] = A[M,K] * W^T + bias[n].
// ---------------------------------------------------------------------------
template <int A_F16, int OUT_F16>
__global__ void __launch_bounds__(32)
fc_wmma_k(const _Float16* __restrict__ Ah, const float* __restrict__ Af,
          const float* __restrict__ Wt, const float* __restrict__ bias,
          _Float16* __restrict__ out_h, float* __restrict__ out_f,
          int N, int K, int ntsh) {
  const int mt = blockIdx.x >> ntsh;
  const int nt = blockIdx.x & ((1 << ntsh) - 1);
  const int lane = threadIdx.x & 31;
  const int ml = lane & 15, hi = lane >> 4;
  const size_t arow = (size_t)(mt * 16 + ml) * K;
  const size_t bcol = (size_t)(nt * 16 + ml) * K;

  v8f acc = {};
  for (int ks = 0; ks < K; ks += 32) {
    if (ks + 32 < K) __builtin_prefetch(&Wt[bcol + ks + 32], 0, 3);
    v16h af, bf;
    #pragma unroll
    for (int v = 0; v < 8; ++v) {
      int ka = hi * 16 + 2 * v;
      if (A_F16) {
        af[2 * v]     = Ah[arow + ks + ka];
        af[2 * v + 1] = Ah[arow + ks + ka + 1];
      } else {
        af[2 * v]     = (_Float16)Af[arow + ks + ka];
        af[2 * v + 1] = (_Float16)Af[arow + ks + ka + 1];
      }
      int kb = hi * 16 + 2 * v;
      bf[2 * v]     = (_Float16)Wt[bcol + ks + kb];
      bf[2 * v + 1] = (_Float16)Wt[bcol + ks + kb + 1];
    }
    acc = __builtin_amdgcn_wmma_f32_16x16x32_f16(false, af, false, bf,
                                                 (short)0, acc, false, false);
  }
  #pragma unroll
  for (int r = 0; r < 8; ++r) {
    int m  = mt * 16 + r + hi * 8;
    int nn = nt * 16 + ml;
    float v = acc[r] + bias[nn];
    if (OUT_F16) out_h[(size_t)m * N + nn] = (_Float16)v;
    else         out_f[(size_t)m * N + nn] = v;
  }
}

// ---------------------------------------------------------------------------
// kappa = softplus(hf . wk + bk) + 1, one block per batch row
// ---------------------------------------------------------------------------
__global__ void __launch_bounds__(256)
kappa_k(const _Float16* __restrict__ hf, const float* __restrict__ wk,
        const float* __restrict__ bk, float* __restrict__ kout, int K) {
  __shared__ float red[256];
  int b = blockIdx.x;
  float s = 0.f;
  for (int k = threadIdx.x; k < K; k += 256)
    s += (float)hf[(size_t)b * K + k] * wk[k];
  red[threadIdx.x] = s;
  __syncthreads();
  for (int off = 128; off; off >>= 1) {
    if (threadIdx.x < off) red[threadIdx.x] += red[threadIdx.x + off];
    __syncthreads();
  }
  if (threadIdx.x == 0) {
    float x  = red[0] + bk[0];
    float sp = (x > 20.f) ? x : log1pf(expf(x));
    kout[b] = sp + 1.f;
  }
}

// ---------------------------------------------------------------------------
// vMF sampling: 1 wave (32 lanes == LATENT) per batch row.
// ---------------------------------------------------------------------------
__device__ __forceinline__ uint32_t pcg_hash(uint32_t v) {
  v = v * 747796405u + 2891336453u;
  uint32_t w = ((v >> ((v >> 28u) + 4u)) ^ v) * 277803737u;
  return (w >> 22u) ^ w;
}
__device__ __forceinline__ float u01(uint32_t s) {
  return ((pcg_hash(s) >> 8) + 0.5f) * (1.0f / 16777216.0f);
}
__device__ __forceinline__ float wave_sum(float v) {
  #pragma unroll
  for (int off = 16; off; off >>= 1) v += __shfl_xor(v, off, 32);
  return v;
}

__global__ void __launch_bounds__(32)
sample_k(const float* __restrict__ mu_raw, const float* __restrict__ kap,
         float* __restrict__ mu_out, float* __restrict__ kap_out,
         float* __restrict__ z_out) {
  const int b = blockIdx.x;
  const int j = threadIdx.x;
  float mv  = mu_raw[b * 32 + j];
  float nrm = fmaxf(sqrtf(wave_sum(mv * mv)), 1e-12f);
  float mu  = mv / nrm;
  float kv  = kap[b];

  const float d  = 32.f;
  float bp = (-2.f * kv + sqrtf(4.f * kv * kv + (d - 1.f) * (d - 1.f))) / (d - 1.f);
  float x0 = (1.f - bp) / (1.f + bp);
  float mm = (d - 1.f) * 0.5f;
  float c2 = kv * x0 + (d - 1.f) * logf(1.f - x0 * x0);

  float w = 0.f; int done = 0;
  for (int it = 0; it < 100; ++it) {
    uint32_t s = (uint32_t)(b * 100 + it) * 2654435761u + 12345u;
    float u1 = u01(s ^ 0x9E3779B9u);
    float u2 = u01(s ^ 0x85EBCA6Bu);
    float u3 = u01(s ^ 0xC2B2AE35u);
    float nz = sqrtf(-2.f * logf(u1)) * cosf(6.28318530718f * u2);
    float zb = fminf(fmaxf(0.5f + nz * rsqrtf(8.f * mm + 4.f), 1e-6f), 1.f - 1e-6f);
    float Wp = (1.f - (1.f + bp) * zb) / (1.f - (1.f - bp) * zb);
    int accp = logf(u3) <= kv * Wp + (d - 1.f) * logf(1.f - x0 * Wp) - c2;
    if (accp && !done) { w = Wp; done = 1; }
  }

  uint32_t se = (uint32_t)(b * 32 + j) * 2246822519u + 777u;
  float e1 = u01(se ^ 0xA511E9B3u), e2 = u01(se ^ 0xB4B82E39u);
  float eps = sqrtf(-2.f * logf(e1)) * cosf(6.28318530718f * e2);

  float dot = wave_sum(eps * mu);
  float ev  = eps - dot * mu;
  float en  = fmaxf(sqrtf(wave_sum(ev * ev)), 1e-12f);
  ev /= en;
  float zv = w * mu + sqrtf(fmaxf(1.f - w * w, 1e-10f)) * ev;
  float zn = fmaxf(sqrtf(wave_sum(zv * zv)), 1e-12f);
  zv /= zn;

  mu_out[b * 32 + j] = mu;
  z_out[b * 32 + j]  = zv;
  if (j == 0) kap_out[b] = kv;
}

// ---------------------------------------------------------------------------
extern "C" void kernel_launch(void* const* d_in, const int* in_sizes, int n_in,
                              void* d_out, int out_size, void* d_ws, size_t ws_size,
                              hipStream_t stream) {
  const float* x   = (const float*)d_in[0];
  const float* w1  = (const float*)d_in[1];  const float* b1   = (const float*)d_in[2];
  const float* g1  = (const float*)d_in[3];  const float* be1  = (const float*)d_in[4];
  const float* m1  = (const float*)d_in[5];  const float* v1   = (const float*)d_in[6];
  const float* w2  = (const float*)d_in[7];  const float* b2   = (const float*)d_in[8];
  const float* g2  = (const float*)d_in[9];  const float* be2  = (const float*)d_in[10];
  const float* m2  = (const float*)d_in[11]; const float* v2   = (const float*)d_in[12];
  const float* w3  = (const float*)d_in[13]; const float* b3   = (const float*)d_in[14];
  const float* g3  = (const float*)d_in[15]; const float* be3  = (const float*)d_in[16];
  const float* m3  = (const float*)d_in[17]; const float* v3   = (const float*)d_in[18];
  const float* wmu = (const float*)d_in[19]; const float* bmu  = (const float*)d_in[20];
  const float* wk  = (const float*)d_in[21]; const float* bk   = (const float*)d_in[22];
  const float* wd  = (const float*)d_in[23]; const float* bd   = (const float*)d_in[24];
  const float* dw1 = (const float*)d_in[25]; const float* db1  = (const float*)d_in[26];
  const float* dg1 = (const float*)d_in[27]; const float* dbe1 = (const float*)d_in[28];
  const float* dm1 = (const float*)d_in[29]; const float* dv1  = (const float*)d_in[30];
  const float* dw2 = (const float*)d_in[31]; const float* db2  = (const float*)d_in[32];
  const float* dg2 = (const float*)d_in[33]; const float* dbe2 = (const float*)d_in[34];
  const float* dm2 = (const float*)d_in[35]; const float* dv2  = (const float*)d_in[36];
  const float* dw3 = (const float*)d_in[37]; const float* db3  = (const float*)d_in[38];

  const int RECON = 128 * 3 * 128 * 128;   // 6291456

  uint8_t* ws = (uint8_t*)d_ws;
  size_t off = 0;
  auto carve = [&](size_t bytes) -> void* {
    void* p = ws + off;
    off = (off + bytes + 255) & ~(size_t)255;
    return p;
  };
  _Float16* xh   = (_Float16*)carve((size_t)RECON * 2);
  _Float16* h1   = (_Float16*)carve((size_t)128 * 32 * 64 * 64 * 2);
  _Float16* h2   = (_Float16*)carve((size_t)128 * 64 * 32 * 32 * 2);
  _Float16* h3   = (_Float16*)carve((size_t)128 * 128 * 16 * 16 * 2); // hf
  _Float16* hd   = (_Float16*)carve((size_t)128 * 32768 * 2);
  _Float16* t1   = (_Float16*)carve((size_t)128 * 64 * 32 * 32 * 2);
  _Float16* t2   = (_Float16*)carve((size_t)128 * 32 * 64 * 64 * 2);
  _Float16* wh1  = (_Float16*)carve((size_t)32 * 27 * 2);
  _Float16* wh2  = (_Float16*)carve((size_t)64 * 288 * 2);
  _Float16* wh3  = (_Float16*)carve((size_t)128 * 576 * 2);
  _Float16* wdh1 = (_Float16*)carve((size_t)128 * 64 * 9 * 2);
  _Float16* wdh2 = (_Float16*)carve((size_t)64 * 32 * 9 * 2);
  _Float16* wdh3 = (_Float16*)carve((size_t)32 * 3 * 9 * 2);
  float*    mu_raw = (float*)carve((size_t)128 * 32 * 4);
  float*    kapb   = (float*)carve((size_t)128 * 4);

  float* out     = (float*)d_out;          // recon (bilinear resize == identity)
  float* mu_out  = out + RECON;
  float* kap_out = mu_out + 128 * 32;
  float* z_out   = kap_out + 128;

  // f16 activations + f16 weights (TDM payload)
  cast_h_k<<<(RECON + 255) / 256, 256, 0, stream>>>(x, xh, RECON);
  cast_h_k<<<(32 * 27 + 255) / 256, 256, 0, stream>>>(w1, wh1, 32 * 27);
  cast_h_k<<<(64 * 288 + 255) / 256, 256, 0, stream>>>(w2, wh2, 64 * 288);
  cast_h_k<<<(128 * 576 + 255) / 256, 256, 0, stream>>>(w3, wh3, 128 * 576);
  cast_h_k<<<(128 * 64 * 9 + 255) / 256, 256, 0, stream>>>(dw1, wdh1, 128 * 64 * 9);
  cast_h_k<<<(64 * 32 * 9 + 255) / 256, 256, 0, stream>>>(dw2, wdh2, 64 * 32 * 9);
  cast_h_k<<<(32 * 3 * 9 + 255) / 256, 256, 0, stream>>>(dw3, wdh3, 32 * 3 * 9);

  // encoder: conv s2 + BN + lrelu (implicit-GEMM WMMA; TDM weight staging)
  conv_wmma_k<0, 0, 0><<<dim3(2048, 2), 256, 0, stream>>>(xh, wh1, b1, g1, be1, m1, v1,
      h1, nullptr, 3, 32, /*osh*/6, /*ish*/7);
  conv_wmma_k<0, 0, 1><<<dim3(512, 4), 256, 0, stream>>>(h1, wh2, b2, g2, be2, m2, v2,
      h2, nullptr, 32, 64, 5, 6);
  conv_wmma_k<0, 0, 1><<<dim3(128, 8), 256, 0, stream>>>(h2, wh3, b3, g3, be3, m3, v3,
      h3, nullptr, 64, 128, 4, 5);

  // latent head
  fc_wmma_k<1, 0><<<16, 32, 0, stream>>>(h3, nullptr, wmu, bmu,
      nullptr, mu_raw, 32, 32768, /*ntsh*/1);
  kappa_k<<<128, 256, 0, stream>>>(h3, wk, bk, kapb, 32768);
  sample_k<<<128, 32, 0, stream>>>(mu_raw, kapb, mu_out, kap_out, z_out);

  // decoder FC: hd = z @ wd^T + bd
  fc_wmma_k<0, 1><<<16384, 32, 0, stream>>>(nullptr, z_out, wd, bd,
      hd, nullptr, 32768, 32, /*ntsh*/11);

  // decoder: convT s2 + BN + lrelu; final convT + sigmoid -> d_out
  conv_wmma_k<1, 0, 1><<<dim3(512, 4), 256, 0, stream>>>(hd, wdh1, db1, dg1, dbe1, dm1,
      dv1, t1, nullptr, 128, 64, 5, 4);
  conv_wmma_k<1, 0, 1><<<dim3(2048, 2), 256, 0, stream>>>(t1, wdh2, db2, dg2, dbe2, dm2,
      dv2, t2, nullptr, 64, 32, 6, 5);
  conv_wmma_k<1, 1, 1><<<dim3(8192, 1), 256, 0, stream>>>(t2, wdh3, db3, nullptr, nullptr,
      nullptr, nullptr, nullptr, out, 32, 3, 7, 6);
}
